// GADGNN_32701880991952
// MI455X (gfx1250) — matrix operand order repaired and checked by reading, not compiled, not run on hardware
//
#include <hip/hip_runtime.h>
#include <hip/hip_bf16.h>

typedef __attribute__((ext_vector_type(16))) __bf16 v16bf;
typedef __attribute__((ext_vector_type(8)))  float  v8f;

#define N_NODES 100000
#define FDIM    128
#define NEDGE   600000
#define NGRAPH  1000
#define CWIDTH  4
#define HD      256
#define EPSV    1e-5f

#define GF_ACCUM 1
#define GF_ACT   2

// ---------------- elementwise / graph helpers ----------------

__global__ void k_zero(float* p, long long n) {
  long long i = (long long)blockIdx.x * blockDim.x + threadIdx.x;
  if (i < n) p[i] = 0.0f;
}

__global__ void k_deg(const int* __restrict__ src, float* deg, int E) {
  int e = blockIdx.x * blockDim.x + threadIdx.x;
  if (e < E) atomicAdd(&deg[src[e]], 1.0f);
}

__device__ __forceinline__ float disv(float d) {
  return d > 0.0f ? rsqrtf(fmaxf(d, 1.0f)) : 0.0f;
}

__global__ void k_ew(const int* __restrict__ src, const int* __restrict__ dst,
                     const float* __restrict__ deg, float* ew, int E) {
  int e = blockIdx.x * blockDim.x + threadIdx.x;
  if (e < E) ew[e] = -disv(deg[src[e]]) * disv(deg[dst[e]]);
}

// scatter: Out[dst] += scale * ew[e] * X[src]   (4 floats per thread)
__global__ void k_spmm(const int* __restrict__ src, const int* __restrict__ dst,
                       const float* __restrict__ ew, const float* __restrict__ X,
                       float* __restrict__ O, int E, float scale) {
  long long t = (long long)blockIdx.x * blockDim.x + threadIdx.x;
  int e = (int)(t >> 5);
  if (e >= E) return;
  int f = (int)(t & 31) * 4;
  int s = src[e], d = dst[e];
  float w = ew[e] * scale;
  const float4 x = *(const float4*)(X + (long long)s * FDIM + f);
  float* o = O + (long long)d * FDIM + f;
  atomicAdd(o + 0, w * x.x);
  atomicAdd(o + 1, w * x.y);
  atomicAdd(o + 2, w * x.z);
  atomicAdd(o + 3, w * x.w);
}

__global__ void k_sub(float* __restrict__ a, const float* __restrict__ b, long long n) {
  long long i = (long long)blockIdx.x * blockDim.x + threadIdx.x;
  if (i < n) a[i] = a[i] - b[i];
}

// ---------------- WMMA operand packing (f32 -> bf16, contiguous-k lane layout) ----------------

// A[M,K] row-major -> tiles (mt,kt), per tile 32 lanes x v16bf.
// Lane L: row = mt*16 + (L&15); elem i -> k = kt*32+(L>>4)*16 + i.
__global__ void k_packA(const float* __restrict__ A, v16bf* __restrict__ P,
                        int M, int K, int Mt, int Kt) {
  int t = blockIdx.x * blockDim.x + threadIdx.x;
  int total = Mt * Kt * 32;
  if (t >= total) return;
  int lane = t & 31;
  int tile = t >> 5;
  int kt = tile % Kt, mt = tile / Kt;
  int row = mt * 16 + (lane & 15);
  int kb  = kt * 32 + ((lane >> 4) * 16);
  v16bf v;
  if (row < M) {
    const float* a = A + (long long)row * K;
#pragma unroll
    for (int i = 0; i < 8; ++i) v[i]     = (__bf16)a[kb + i];
#pragma unroll
    for (int i = 0; i < 8; ++i) v[8 + i] = (__bf16)a[kb + 8 + i];
  } else {
#pragma unroll
    for (int i = 0; i < 16; ++i) v[i] = (__bf16)0.0f;
  }
  P[t] = v;
}

// W[K,Nc] row-major -> tiles (kt,nt). Lane L: col = nt*16 + (L&15);
// elem i -> k = kt*32 + (L>>4)*16 + i.
__global__ void k_packB(const float* __restrict__ W, v16bf* __restrict__ P,
                        int K, int Nc, int Kt, int Nt) {
  int t = blockIdx.x * blockDim.x + threadIdx.x;
  int total = Kt * Nt * 32;
  if (t >= total) return;
  int lane = t & 31;
  int tile = t >> 5;
  int nt = tile % Nt, kt = tile / Nt;
  int n  = nt * 16 + (lane & 15);
  int k0 = kt * 32 + ((lane >> 4) * 16);
  v16bf v;
#pragma unroll
  for (int i = 0; i < 16; ++i) {
    float x = (n < Nc) ? W[(long long)(k0 + i) * Nc + n] : 0.0f;
    v[i] = (__bf16)x;
  }
  P[t] = v;
}

// ---------------- WMMA GEMM: one wave per 16x128 C strip (8 N-tiles) ----------------
// B panel preloaded into registers per k-step (single wait, 8 back-to-back WMMAs);
// A fragment software-pipelined one k-step ahead.
// C[M, ldc] (f32), strip nb covers cols [colOff + nb*128, colOff + nb*128 + 128)

__global__ __launch_bounds__(256)
void k_gemm(const v16bf* __restrict__ Ap, const v16bf* __restrict__ Bp,
            const float* __restrict__ bias, float* __restrict__ C,
            int M, int Mt, int Nb, int Nt, int Kt, int ldc, int colOff, int flags) {
  int wave = (int)((blockIdx.x * blockDim.x + threadIdx.x) >> 5);
  int lane = threadIdx.x & 31;
  if (wave >= Mt * Nb) return;
  int nb = wave % Nb, mt = wave / Nb;

  v8f acc[8] = {};
  const v16bf* ap = Ap + (long long)mt * Kt * 32 + lane;
  const v16bf* bp = Bp + ((long long)nb * 8) * 32 + lane;

  v16bf a = ap[0];
  for (int kt = 0; kt < Kt; ++kt) {
    const v16bf* bk = bp + (long long)kt * Nt * 32;
    v16bf b[8];
#pragma unroll
    for (int j = 0; j < 8; ++j) b[j] = bk[j * 32];     // one clause of loads
    v16bf an = ap[(kt + 1) * 32];                      // pipelined next-A (in-ws overread ok)
    __builtin_prefetch(ap + (kt + 2) * 32, 0, 1);      // speculative prefetch
#pragma unroll
    for (int j = 0; j < 8; ++j) {                      // 8 back-to-back WMMAs
      acc[j] = __builtin_amdgcn_wmma_f32_16x16x32_bf16(false, a, false, b[j],
                                                       (short)0, acc[j], false, false);
    }
    a = an;
  }

  int nl0 = nb * 128 + (lane & 15);
#pragma unroll
  for (int j = 0; j < 8; ++j) {
    int nl  = nl0 + j * 16;
    int col = colOff + nl;
    float bv = bias ? bias[nl] : 0.0f;
#pragma unroll
    for (int g = 0; g < 8; ++g) {
      int row = mt * 16 + ((lane >> 4) * 8 + g);
      if (row < M) {
        float* cp = C + (long long)row * ldc + col;
        float v = acc[j][g] + bv;
        if (flags & GF_ACCUM) v += *cp;
        if (flags & GF_ACT)   v = v > 0.0f ? v : 0.01f * v;
        *cp = v;
      }
    }
  }
}

// ---------------- attention pooling / BN / head ----------------

__global__ void k_scores(const float* __restrict__ h, const float* __restrict__ tmp,
                         const int* __restrict__ gid, float* __restrict__ scores, int n) {
  int wave = (int)((blockIdx.x * blockDim.x + threadIdx.x) >> 5);
  int lane = threadIdx.x & 31;
  if (wave >= n) return;
  int g = gid[wave];
  const float* hv = h + (long long)wave * HD;
  const float* tv = tmp + (long long)g * HD;
  float s = 0.0f;
  for (int j = lane; j < HD; j += 32) s += hv[j] * tv[j];
#pragma unroll
  for (int off = 16; off; off >>= 1) s += __shfl_down(s, off, 32);
  if (lane == 0) scores[wave] = s;
}

__global__ void k_count(const int* __restrict__ gid, float* counts, int n) {
  int v = blockIdx.x * blockDim.x + threadIdx.x;
  if (v < n) atomicAdd(&counts[gid[v]], 1.0f);
}

// hc[g, 0:256] += h[v] * scores[v]/max(count,1)
__global__ void k_pool(const float* __restrict__ h, const float* __restrict__ scores,
                       const float* __restrict__ counts, const int* __restrict__ gid,
                       float* __restrict__ hc, int n) {
  long long t = (long long)blockIdx.x * blockDim.x + threadIdx.x;
  int v = (int)(t >> 6);
  if (v >= n) return;
  int f = (int)(t & 63) * 4;
  int g = gid[v];
  float w = scores[v] / fmaxf(counts[g], 1.0f);
  const float4 x = *(const float4*)(h + (long long)v * HD + f);
  float* o = hc + (long long)g * (2 * HD) + f;
  atomicAdd(o + 0, w * x.x);
  atomicAdd(o + 1, w * x.y);
  atomicAdd(o + 2, w * x.z);
  atomicAdd(o + 3, w * x.w);
}

__global__ void k_bnstat(const float* __restrict__ hc, float* mu, float* var) {
  __shared__ float s1[256], s2[256];
  int j = blockIdx.x;            // 0..511
  int tid = threadIdx.x;
  float a = 0.0f, b = 0.0f;
  for (int g = tid; g < NGRAPH; g += 256) {
    float x = hc[(long long)g * (2 * HD) + j];
    a += x; b += x * x;
  }
  s1[tid] = a; s2[tid] = b; __syncthreads();
  for (int off = 128; off; off >>= 1) {
    if (tid < off) { s1[tid] += s1[tid + off]; s2[tid] += s2[tid + off]; }
    __syncthreads();
  }
  if (tid == 0) {
    float m = s1[0] / (float)NGRAPH;
    mu[j] = m;
    var[j] = s2[0] / (float)NGRAPH - m * m;
  }
}

__global__ void k_final(const float* __restrict__ hc, const float* __restrict__ mu,
                        const float* __restrict__ var, const float* __restrict__ gamma,
                        const float* __restrict__ beta, const float* __restrict__ W7,
                        const float* __restrict__ b7, float* __restrict__ out) {
  int g = blockIdx.x * blockDim.x + threadIdx.x;
  if (g >= NGRAPH) return;
  float a0 = b7[0], a1 = b7[1];
  for (int j = 0; j < 2 * HD; ++j) {
    float xn = gamma[j] * (hc[(long long)g * (2 * HD) + j] - mu[j]) *
               rsqrtf(var[j] + EPSV) + beta[j];
    a0 += xn * W7[j * 2 + 0];
    a1 += xn * W7[j * 2 + 1];
  }
  out[g * 2 + 0] = a0;
  out[g * 2 + 1] = a1;
}

// ---------------- host orchestration ----------------

static inline int divUp(long long a, int b) { return (int)((a + b - 1) / b); }

extern "C" void kernel_launch(void* const* d_in, const int* in_sizes, int n_in,
                              void* d_out, int out_size, void* d_ws, size_t ws_size,
                              hipStream_t stream) {
  const float* X    = (const float*)d_in[0];
  const int*   ei   = (const int*)d_in[1];
  const int*   gid  = (const int*)d_in[2];
  const float* xlx  = (const float*)d_in[3];
  const float* W1 = (const float*)d_in[4];  const float* b1 = (const float*)d_in[5];
  const float* W2 = (const float*)d_in[6];  const float* b2 = (const float*)d_in[7];
  const float* chW = (const float*)d_in[8]; const float* chB = (const float*)d_in[9];
  const float* W3 = (const float*)d_in[10]; const float* b3 = (const float*)d_in[11];
  const float* W4 = (const float*)d_in[12]; const float* b4 = (const float*)d_in[13];
  const float* W5 = (const float*)d_in[14]; const float* b5 = (const float*)d_in[15];
  const float* W6 = (const float*)d_in[16]; const float* b6 = (const float*)d_in[17];
  const float* W8 = (const float*)d_in[18]; const float* b8 = (const float*)d_in[19];
  const float* W9 = (const float*)d_in[20]; const float* b9 = (const float*)d_in[21];
  const float* W7 = (const float*)d_in[22]; const float* b7 = (const float*)d_in[23];
  const float* gamma = (const float*)d_in[24];
  const float* beta  = (const float*)d_in[25];
  const int* src = ei;
  const int* dst = ei + NEDGE;

  char* ws = (char*)d_ws;
  size_t off = 0;
  auto alloc = [&](size_t bytes) -> char* {
    char* p = ws + off;
    off += (bytes + 255) & ~(size_t)255;
    return p;
  };
  float* deg  = (float*)alloc((size_t)N_NODES * 4);
  float* ew   = (float*)alloc((size_t)NEDGE * 4);
  float* h1   = (float*)alloc((size_t)N_NODES * FDIM * 4);
  float* h    = (float*)alloc((size_t)N_NODES * FDIM * 4);
  float* t1   = (float*)alloc((size_t)N_NODES * FDIM * 4);
  float* t2   = (float*)alloc((size_t)N_NODES * FDIM * 4);
  float* hfin = (float*)alloc((size_t)N_NODES * 4 * FDIM * 4);
  float* h3   = (float*)alloc((size_t)N_NODES * HD * 4);
  float* h4   = (float*)alloc((size_t)N_NODES * HD * 4);
  v16bf* pA   = (v16bf*)alloc((size_t)N_NODES * 512 * 2);     // packed activations
  v16bf* pW   = (v16bf*)alloc((size_t)512 * 256 * 2);         // packed weights
  v16bf* pG   = (v16bf*)alloc((size_t)1008 * 256 * 2);        // packed graph acts
  float* scores = (float*)alloc((size_t)N_NODES * 4);
  float* counts = (float*)alloc((size_t)NGRAPH * 4);
  float* hc   = (float*)alloc((size_t)NGRAPH * 2 * HD * 4);
  float* g1   = (float*)alloc((size_t)NGRAPH * HD * 4);
  float* g2   = (float*)alloc((size_t)NGRAPH * HD * 4);
  float* mu   = (float*)alloc(512 * 4);
  float* var  = (float*)alloc(512 * 4);

  auto zero = [&](float* p, long long n) {
    k_zero<<<divUp(n, 256), 256, 0, stream>>>(p, n);
  };
  auto packA = [&](const float* A, v16bf* P, int M, int K) {
    int Mt = (M + 15) / 16, Kt = K / 32;
    k_packA<<<divUp((long long)Mt * Kt * 32, 256), 256, 0, stream>>>(A, P, M, K, Mt, Kt);
  };
  auto packB = [&](const float* Wp, int K, int Nc) {
    int Kt = K / 32, Nt = (Nc + 15) / 16;
    k_packB<<<divUp((long long)Kt * Nt * 32, 256), 256, 0, stream>>>(Wp, pW, K, Nc, Kt, Nt);
  };
  auto gemm = [&](const v16bf* Ap, const float* bias, float* C,
                  int M, int K, int Nc, int ldc, int colOff, int flags) {
    int Mt = (M + 15) / 16, Nt = (Nc + 15) / 16, Kt = K / 32;
    int Nb = Nt / 8;                       // all Nc here are multiples of 128
    long long waves = (long long)Mt * Nb;
    k_gemm<<<divUp(waves * 32, 256), 256, 0, stream>>>(Ap, pW, bias, C, M, Mt, Nb, Nt,
                                                       Kt, ldc, colOff, flags);
  };

  // --- edge weights: deg over src, ew = -dis[src]*dis[dst] ---
  zero(deg, N_NODES);
  k_deg<<<divUp(NEDGE, 256), 256, 0, stream>>>(src, deg, NEDGE);
  k_ew<<<divUp(NEDGE, 256), 256, 0, stream>>>(src, dst, deg, ew, NEDGE);

  // --- h = act(act(X@W1+b1)@W2+b2) ---
  packA(X, pA, N_NODES, FDIM);
  packB(W1, FDIM, FDIM);
  gemm(pA, b1, h1, N_NODES, FDIM, FDIM, FDIM, 0, GF_ACT);
  packA(h1, pA, N_NODES, FDIM);
  packB(W2, FDIM, FDIM);
  gemm(pA, b2, h, N_NODES, FDIM, FDIM, FDIM, 0, GF_ACT);

  // --- Chebyshev basis: t1 = prop(h); t2 = 2*prop(t1) - h (branch independent) ---
  zero(t1, (long long)N_NODES * FDIM);
  k_spmm<<<divUp((long long)NEDGE * 32, 256), 256, 0, stream>>>(src, dst, ew, h, t1, NEDGE, 1.0f);
  zero(t2, (long long)N_NODES * FDIM);
  k_spmm<<<divUp((long long)NEDGE * 32, 256), 256, 0, stream>>>(src, dst, ew, t1, t2, NEDGE, 2.0f);
  k_sub<<<divUp((long long)N_NODES * FDIM, 256), 256, 0, stream>>>(t2, h, (long long)N_NODES * FDIM);

  // --- h_final[:, br*128:(br+1)*128] = h@W[br,0] + t1@W[br,1] + t2@W[br,2] + b[br] ---
  packA(h, pA, N_NODES, FDIM);
  for (int br = 0; br < CWIDTH; ++br) {
    packB(chW + ((size_t)br * 3 + 0) * FDIM * FDIM, FDIM, FDIM);
    gemm(pA, nullptr, hfin, N_NODES, FDIM, FDIM, 4 * FDIM, br * FDIM, 0);
  }
  packA(t1, pA, N_NODES, FDIM);
  for (int br = 0; br < CWIDTH; ++br) {
    packB(chW + ((size_t)br * 3 + 1) * FDIM * FDIM, FDIM, FDIM);
    gemm(pA, nullptr, hfin, N_NODES, FDIM, FDIM, 4 * FDIM, br * FDIM, GF_ACCUM);
  }
  packA(t2, pA, N_NODES, FDIM);
  for (int br = 0; br < CWIDTH; ++br) {
    packB(chW + ((size_t)br * 3 + 2) * FDIM * FDIM, FDIM, FDIM);
    gemm(pA, chB + (size_t)br * FDIM, hfin, N_NODES, FDIM, FDIM, 4 * FDIM, br * FDIM, GF_ACCUM);
  }

  // --- h4 = act(act(hfin@W3+b3)@W4+b4) ---
  packA(hfin, pA, N_NODES, 4 * FDIM);
  packB(W3, 4 * FDIM, HD);
  gemm(pA, b3, h3, N_NODES, 4 * FDIM, HD, HD, 0, GF_ACT);
  packA(h3, pA, N_NODES, HD);
  packB(W4, HD, HD);
  gemm(pA, b4, h4, N_NODES, HD, HD, HD, 0, GF_ACT);

  // --- tmp = act(act(xLx@W8+b8)@W9+b9) (graph side) ---
  packA(xlx, pG, NGRAPH, FDIM);
  packB(W8, FDIM, HD);
  gemm(pG, b8, g1, NGRAPH, FDIM, HD, HD, 0, GF_ACT);
  packA(g1, pG, NGRAPH, HD);
  packB(W9, HD, HD);
  gemm(pG, b9, g2, NGRAPH, HD, HD, HD, 0, GF_ACT);

  // --- attention pooling into hc[:, 0:256] ---
  k_scores<<<divUp((long long)N_NODES * 32, 256), 256, 0, stream>>>(h4, g2, gid, scores, N_NODES);
  zero(counts, NGRAPH);
  k_count<<<divUp(N_NODES, 256), 256, 0, stream>>>(gid, counts, N_NODES);
  zero(hc, (long long)NGRAPH * 2 * HD);
  k_pool<<<divUp((long long)N_NODES * 64, 256), 256, 0, stream>>>(h4, scores, counts, gid, hc, N_NODES);

  // --- xLx path into hc[:, 256:512] ---
  packA(xlx, pG, NGRAPH, FDIM);
  packB(W5, FDIM, HD);
  gemm(pG, b5, g1, NGRAPH, FDIM, HD, HD, 0, GF_ACT);
  packA(g1, pG, NGRAPH, HD);
  packB(W6, HD, HD);
  gemm(pG, b6, hc, NGRAPH, HD, HD, 2 * HD, HD, GF_ACT);

  // --- BatchNorm (batch stats) + head ---
  k_bnstat<<<2 * HD, 256, 0, stream>>>(hc, mu, var);
  k_final<<<divUp(NGRAPH, 128), 128, 0, stream>>>(hc, mu, var, gamma, beta, W7, b7,
                                                  (float*)d_out);
}
